// UMT5Attention_61727269978629
// MI455X (gfx1250) — matrix operand order repaired and checked by reading, not compiled, not run on hardware
//
#include <hip/hip_runtime.h>
#include <hip/hip_bf16.h>

// ---------------------------------------------------------------------------
// UMT5 attention, CDNA5 (gfx1250) WMMA implementation.
// B=2, S=2048, D_MODEL=1024, H=16, D_KV=64, NUM_BUCKETS=32, MAX_DISTANCE=128
//   1) hs f32 -> f16 ; W* f32 -> f16 transposed ; bias table [h][rel+2047]
//   2) Q/K = hs @ W   (WMMA GEMM, f16 scattered to [b][h][s][d])
//      V   = hs @ Wv  (WMMA GEMM, f16 scattered TRANSPOSED to [b][h][d][s])
//   3) flash attention: v_wmma scores + online softmax + v_wmma P*V
//   4) out = ctx @ Wo (WMMA GEMM, f32 out)
// LDS staging uses GLOBAL_LOAD_ASYNC_TO_LDS_B128 (ASYNCcnt-tracked).
// ---------------------------------------------------------------------------

typedef _Float16 h16;
typedef __attribute__((ext_vector_type(16))) _Float16 v16h;
typedef __attribute__((ext_vector_type(8)))  _Float16 v8h;
typedef __attribute__((ext_vector_type(8)))  float    v8f;
typedef __attribute__((ext_vector_type(4)))  int      v4i;

#define S_LEN 2048
#define D_MODEL 1024
#define N_HEADS 16
#define D_KV 64
#define BATCH 2

// ---- async memory->LDS staging (CDNA5 GLOBAL_LOAD_ASYNC_TO_LDS_B128) ------

#if __has_builtin(__builtin_amdgcn_global_load_async_to_lds_b128) && \
    __has_builtin(__builtin_amdgcn_s_wait_asynccnt)
#define USE_ASYNC_COPY 1
#else
#define USE_ASYNC_COPY 0
#endif

typedef __attribute__((address_space(1))) v4i gv4i;   // global b128
typedef __attribute__((address_space(3))) v4i lv4i;   // LDS b128

__device__ __forceinline__ void cp_b128(h16* lds_dst, const h16* gsrc) {
#if USE_ASYNC_COPY
  __builtin_amdgcn_global_load_async_to_lds_b128((gv4i*)gsrc, (lv4i*)lds_dst,
                                                 0, 0);
#else
  *(v8h*)lds_dst = *(const v8h*)gsrc;
#endif
}

__device__ __forceinline__ void cp_wait() {
#if USE_ASYNC_COPY
  __builtin_amdgcn_s_wait_asynccnt(0);
#endif
}

// ---- WMMA fragment helpers (wave32 layouts per CDNA5 ISA 7.12.2) ----------

// A fragment, 16x32 f16, source row-major with leading dim `ld`.
// lane<16 holds row m=lane, K {0..7,16..23}; lane>=16 holds K {8..15,24..31}.
__device__ __forceinline__ v16h frag_a_load(const h16* tile, int ld, int lane) {
  const int m  = lane & 15;
  const int ko = (lane >> 4) * 8;
  const h16* p = tile + (size_t)m * ld + ko;
  v8h lo = *(const v8h*)p;
  v8h hi = *(const v8h*)(p + 16);
  return __builtin_shufflevector(lo, hi, 0,1,2,3,4,5,6,7,8,9,10,11,12,13,14,15);
}

// B fragment, 32x16 f16 (KxN), source stored TRANSPOSED (N rows x K cols, ld).
// lane n=lane&15, K {0..7,16..23} for lane<16, K {8..15,24..31} for lane>=16.
__device__ __forceinline__ v16h frag_b_load(const h16* bt, int ld, int kbase, int lane) {
  const int n  = lane & 15;
  const int kb = (lane >> 4) * 8;
  const h16* p = bt + (size_t)n * ld + kbase + kb;
  v8h lo = *(const v8h*)p;
  v8h hi = *(const v8h*)(p + 16);
  return __builtin_shufflevector(lo, hi, 0,1,2,3,4,5,6,7,8,9,10,11,12,13,14,15);
}

__device__ __forceinline__ v8f wmma_f16(v16h a, v16h b, v8f c) {
  return __builtin_amdgcn_wmma_f32_16x16x32_f16(false, a, false, b, (short)0, c,
                                                false, false);
}

// ---- prep kernels ---------------------------------------------------------

__global__ __launch_bounds__(256) void f32_to_f16_kernel(
    const float* __restrict__ in, h16* __restrict__ out, int n) {
  int i = blockIdx.x * 256 + threadIdx.x;
  if (i < n) out[i] = (h16)in[i];
}

// Wt[n][k] = (f16) W[k][n]  for a 1024x1024 matrix.
__global__ __launch_bounds__(256) void transpose_w_kernel(
    const float* __restrict__ W, h16* __restrict__ Wt) {
  int idx = blockIdx.x * 256 + threadIdx.x;   // [0, 1M)
  int k = idx >> 10, n = idx & 1023;
  Wt[(size_t)n * D_MODEL + k] = (h16)W[idx];
}

// bias table: tab[h][rel + 2047], rel = k - q in [-2047, 2047]
__global__ __launch_bounds__(256) void bias_table_kernel(
    const float* __restrict__ rel_bias, float* __restrict__ tab) {
  int idx = blockIdx.x * 256 + threadIdx.x;
  if (idx >= N_HEADS * (2 * S_LEN - 1)) return;
  int h = idx / (2 * S_LEN - 1);
  int ridx = idx - h * (2 * S_LEN - 1);
  int rel = ridx - (S_LEN - 1);       // mem - ctx
  int n = -rel;
  int ret = (n < 0) ? 16 : 0;         // num_buckets/2
  n = (n < 0) ? -n : n;
  int bucket;
  if (n < 8) {                        // max_exact = 8
    bucket = n;
  } else {
    int v = 8 + (int)(logf((float)n * 0.125f) * (8.0f / logf(16.0f)));
    bucket = (v < 15) ? v : 15;
  }
  bucket += ret;
  tab[(size_t)h * (2 * S_LEN - 1) + ridx] = rel_bias[bucket * N_HEADS + h];
}

// ---- generic WMMA GEMM: C = A(MxK) * Bt^T, A f16 row-major, Bt f16 (NxK) --
// Block tile 256(M) x 64(N), K-step 32. 8 waves, each owns a 32x64 strip
// (2 A-fragments, 8 accumulators -> 8 WMMAs per K-step per wave).
// MODE 0: f16 output scattered to [b][h][s][d]   (Q, K)
// MODE 1: f16 output scattered to [b][h][d][s]   (V, pre-transposed)
// MODE 2: f32 output row-major                   (final projection)

template <int MODE>
__global__ __launch_bounds__(256) void gemm_wmma(
    const h16* __restrict__ A, const h16* __restrict__ Bt,
    void* __restrict__ outp, int Kd, int N) {
  constexpr int BM = 256, BN = 64, BK = 32, LDT = BK + 8;
  __shared__ __align__(16) h16 sA[BM * LDT];   // 20 KB
  __shared__ __align__(16) h16 sB[BN * LDT];   //  5 KB
  const int tid = threadIdx.x, lane = tid & 31, w = tid >> 5;
  const int m0 = blockIdx.x * BM, n0 = blockIdx.y * BN;

  v8f acc[2][4];
  const v8f zero8 = {0.f,0.f,0.f,0.f,0.f,0.f,0.f,0.f};
#pragma unroll
  for (int ms = 0; ms < 2; ++ms)
#pragma unroll
    for (int i = 0; i < 4; ++i) acc[ms][i] = zero8;

  for (int k0 = 0; k0 < Kd; k0 += BK) {
    __syncthreads();
#pragma unroll
    for (int i = 0; i < 4; ++i) {        // A tile: 256x32 = 1024 8-half chunks
      int c = tid + i * 256;
      int r = c >> 2, col = (c & 3) * 8;
      cp_b128(sA + r * LDT + col, A + (size_t)(m0 + r) * Kd + k0 + col);
    }
    {                                    // B tile: 64x32 = 256 chunks
      int r = tid >> 2, col = (tid & 3) * 8;
      cp_b128(sB + r * LDT + col, Bt + (size_t)(n0 + r) * Kd + k0 + col);
    }
    cp_wait();
    __syncthreads();
    v16h a0 = frag_a_load(sA + (w * 32) * LDT, LDT, lane);
    v16h a1 = frag_a_load(sA + (w * 32 + 16) * LDT, LDT, lane);
#pragma unroll
    for (int ns = 0; ns < 4; ++ns) {
      v16h bf = frag_b_load(sB + ns * 16 * LDT, LDT, 0, lane);
      acc[0][ns] = wmma_f16(a0, bf, acc[0][ns]);
      acc[1][ns] = wmma_f16(a1, bf, acc[1][ns]);
    }
  }

  const int mlo = (lane >> 4) * 8, nlan = lane & 15;
#pragma unroll
  for (int ms = 0; ms < 2; ++ms) {
    const int mtop = m0 + w * 32 + ms * 16;
#pragma unroll
    for (int ns = 0; ns < 4; ++ns) {
#pragma unroll
      for (int r = 0; r < 8; ++r) {
        int gr = mtop + mlo + r;
        int gc = n0 + ns * 16 + nlan;
        float val = acc[ms][ns][r];
        if (MODE == 0) {
          int bb = gr >> 11, ss = gr & (S_LEN - 1);
          int hh = gc >> 6, dd = gc & (D_KV - 1);
          ((h16*)outp)[(((size_t)bb * N_HEADS + hh) * S_LEN + ss) * D_KV + dd] =
              (h16)val;
        } else if (MODE == 1) {
          int bb = gr >> 11, ss = gr & (S_LEN - 1);
          int hh = gc >> 6, dd = gc & (D_KV - 1);
          ((h16*)outp)[(((size_t)bb * N_HEADS + hh) * D_KV + dd) * S_LEN + ss] =
              (h16)val;
        } else {
          ((float*)outp)[(size_t)gr * N + gc] = val;
        }
      }
    }
  }
}

// ---- flash attention ------------------------------------------------------
// grid (S/64, H, B), 128 threads (4 waves). Wave w: 16 query rows.
// Streams 64-key chunks of K ([key][d]) and pre-transposed V ([d][key]).

__global__ __launch_bounds__(128) void attn_wmma(
    const h16* __restrict__ Q, const h16* __restrict__ Km,
    const h16* __restrict__ Vt, const float* __restrict__ biasTab,
    const unsigned char* __restrict__ mask, h16* __restrict__ ctx) {
  constexpr int LD = 72;
  __shared__ __align__(16) h16 sK[64 * LD];
  __shared__ __align__(16) h16 sV[64 * LD];
  __shared__ __align__(16) h16 sP[4 * 16 * LD];
  const int tid = threadIdx.x, lane = tid & 31, w = tid >> 5;
  const int b = blockIdx.z, h = blockIdx.y;
  const int q0 = blockIdx.x * 64 + w * 16;
  const size_t headoff = ((size_t)(b * N_HEADS + h)) * S_LEN * D_KV;
  const h16* Qh = Q + headoff;
  const h16* Kh = Km + headoff;
  const h16* Vth = Vt + headoff;                 // [d][s] per head

  const v16h aQ0 = frag_a_load(Qh + (size_t)q0 * D_KV, D_KV, lane);
  const v16h aQ1 = frag_a_load(Qh + (size_t)q0 * D_KV + 32, D_KV, lane);

  const v8f zero8 = {0.f,0.f,0.f,0.f,0.f,0.f,0.f,0.f};
  float mrun[8], lrun[8];
  v8f acc[4];
#pragma unroll
  for (int r = 0; r < 8; ++r) { mrun[r] = -3.0e38f; lrun[r] = 0.f; }
#pragma unroll
  for (int i = 0; i < 4; ++i) acc[i] = zero8;
  const int mlo = (lane >> 4) * 8, nlan = lane & 15;

  for (int kb0 = 0; kb0 < S_LEN; kb0 += 64) {
    __syncthreads();
    // K chunk: sK[key][d]; V chunk: sV[d][key]; 512 8-half chunks each
#pragma unroll
    for (int i = 0; i < 4; ++i) {
      int c = tid + i * 128;
      int r = c >> 3, col = (c & 7) * 8;
      cp_b128(sK + r * LD + col, Kh + (size_t)(kb0 + r) * D_KV + col);
      cp_b128(sV + r * LD + col, Vth + (size_t)r * S_LEN + kb0 + col);
    }
    cp_wait();
    __syncthreads();

    // scores: S = Q (16x64) * K^T -> 16x64 tile in 4 C-fragments
    v8f s[4];
#pragma unroll
    for (int ns = 0; ns < 4; ++ns) {
      v16h b0 = frag_b_load(sK + ns * 16 * LD, LD, 0, lane);
      v16h b1 = frag_b_load(sK + ns * 16 * LD, LD, 32, lane);
      v8f z = zero8;
      z = wmma_f16(aQ0, b0, z);
      z = wmma_f16(aQ1, b1, z);
      s[ns] = z;
    }

    // bias + mask (mask passed as 1-byte bool per numpy bool_)
#pragma unroll
    for (int ns = 0; ns < 4; ++ns) {
      int kg = kb0 + ns * 16 + nlan;
      bool live = mask[(size_t)b * S_LEN + kg] != 0;
      const float* bt = biasTab + (size_t)h * (2 * S_LEN - 1) + kg + (S_LEN - 1);
#pragma unroll
      for (int r = 0; r < 8; ++r) {
        int qg = q0 + mlo + r;
        float sv = s[ns][r];
        s[ns][r] = live ? (sv + bt[-qg]) : -3.0e38f;
      }
    }

    // online softmax update (row reduction across the 16-lane half)
#pragma unroll
    for (int r = 0; r < 8; ++r) {
      float mx = fmaxf(fmaxf(s[0][r], s[1][r]), fmaxf(s[2][r], s[3][r]));
      mx = fmaxf(mx, __shfl_xor(mx, 1, 32));
      mx = fmaxf(mx, __shfl_xor(mx, 2, 32));
      mx = fmaxf(mx, __shfl_xor(mx, 4, 32));
      mx = fmaxf(mx, __shfl_xor(mx, 8, 32));
      float mnew = fmaxf(mrun[r], mx);
      float scale = __expf(mrun[r] - mnew);
      mrun[r] = mnew;
      lrun[r] *= scale;
#pragma unroll
      for (int ds = 0; ds < 4; ++ds) acc[ds][r] *= scale;
      float rs = 0.f;
#pragma unroll
      for (int ns = 0; ns < 4; ++ns) {
        float p = __expf(s[ns][r] - mnew);
        s[ns][r] = p;
        rs += p;
      }
      rs += __shfl_xor(rs, 1, 32);
      rs += __shfl_xor(rs, 2, 32);
      rs += __shfl_xor(rs, 4, 32);
      rs += __shfl_xor(rs, 8, 32);
      lrun[r] += rs;
    }

    // P (C-layout) -> per-wave LDS -> A-layout fragments, then O += P*V
    h16* pbase = sP + w * 16 * LD;
#pragma unroll
    for (int ns = 0; ns < 4; ++ns)
#pragma unroll
      for (int r = 0; r < 8; ++r)
        pbase[(mlo + r) * LD + ns * 16 + nlan] = (h16)s[ns][r];
    v16h aP0 = frag_a_load(pbase, LD, lane);
    v16h aP1 = frag_a_load(pbase + 32, LD, lane);
#pragma unroll
    for (int ds = 0; ds < 4; ++ds) {
      v16h b0 = frag_b_load(sV + ds * 16 * LD, LD, 0, lane);
      v16h b1 = frag_b_load(sV + ds * 16 * LD, LD, 32, lane);
      acc[ds] = wmma_f16(aP0, b0, acc[ds]);
      acc[ds] = wmma_f16(aP1, b1, acc[ds]);
    }
  }

  // epilogue: ctx[b][s][h*64+d] = O / l
#pragma unroll
  for (int ds = 0; ds < 4; ++ds) {
#pragma unroll
    for (int r = 0; r < 8; ++r) {
      int qg = q0 + mlo + r;
      float o = acc[ds][r] / lrun[r];
      ctx[((size_t)(b * S_LEN + qg)) * D_MODEL + h * D_KV + ds * 16 + nlan] =
          (h16)o;
    }
  }
}

// ---- host launcher --------------------------------------------------------

extern "C" void kernel_launch(void* const* d_in, const int* in_sizes, int n_in,
                              void* d_out, int out_size, void* d_ws,
                              size_t ws_size, hipStream_t stream) {
  const float* hs = (const float*)d_in[0];
  const unsigned char* mask = (const unsigned char*)d_in[1];  // jnp.bool_
  const float* Wq = (const float*)d_in[2];
  const float* Wk = (const float*)d_in[3];
  const float* Wv = (const float*)d_in[4];
  const float* Wo = (const float*)d_in[5];
  const float* rb = (const float*)d_in[6];

  char* ws = (char*)d_ws;
  const size_t MB = 1024ull * 1024ull;
  h16* hsH = (h16*)(ws);                 //  8 MB: [4096][1024] f16
  h16* WqT = (h16*)(ws + 8 * MB);        //  2 MB each, transposed [n][k]
  h16* WkT = (h16*)(ws + 10 * MB);
  h16* WvT = (h16*)(ws + 12 * MB);
  h16* WoT = (h16*)(ws + 14 * MB);
  h16* Qb  = (h16*)(ws + 16 * MB);       //  8 MB: [b][h][s][d] f16
  h16* Kb  = (h16*)(ws + 24 * MB);
  h16* Vb  = (h16*)(ws + 32 * MB);       //  8 MB: [b][h][d][s] f16 (transposed)
  h16* ctxH = (h16*)(ws + 40 * MB);      //  8 MB: [b][s][1024] f16
  float* biasTab = (float*)(ws + 48 * MB);  // 16*4095 f32

  const int HS_N = BATCH * S_LEN * D_MODEL;  // 4 Mi elements
  f32_to_f16_kernel<<<HS_N / 256, 256, 0, stream>>>(hs, hsH, HS_N);
  transpose_w_kernel<<<(D_MODEL * D_MODEL) / 256, 256, 0, stream>>>(Wq, WqT);
  transpose_w_kernel<<<(D_MODEL * D_MODEL) / 256, 256, 0, stream>>>(Wk, WkT);
  transpose_w_kernel<<<(D_MODEL * D_MODEL) / 256, 256, 0, stream>>>(Wv, WvT);
  transpose_w_kernel<<<(D_MODEL * D_MODEL) / 256, 256, 0, stream>>>(Wo, WoT);
  bias_table_kernel<<<(N_HEADS * (2 * S_LEN - 1) + 255) / 256, 256, 0, stream>>>(
      rb, biasTab);

  dim3 gg((BATCH * S_LEN) / 256, D_MODEL / 64);  // (16, 16)
  gemm_wmma<0><<<gg, 256, 0, stream>>>(hsH, WqT, Qb, D_MODEL, D_MODEL);
  gemm_wmma<0><<<gg, 256, 0, stream>>>(hsH, WkT, Kb, D_MODEL, D_MODEL);
  gemm_wmma<1><<<gg, 256, 0, stream>>>(hsH, WvT, Vb, D_MODEL, D_MODEL);

  attn_wmma<<<dim3(S_LEN / 64, N_HEADS, BATCH), 128, 0, stream>>>(
      Qb, Kb, Vb, biasTab, mask, ctxH);

  gemm_wmma<2><<<gg, 256, 0, stream>>>(ctxH, WoT, d_out, D_MODEL, D_MODEL);
}
